// MyCrossAttention_72894184948103
// MI455X (gfx1250) — matrix operand order repaired and checked by reading, not compiled, not run on hardware
//
#include <hip/hip_runtime.h>

// Problem constants (match reference)
#define B_   2
#define N_   2048
#define NS_  2048
#define C_   512
#define H_   8
#define HD_  64
#define SCALE_ 0.125f      // HD^-0.5
#define NEG_  -10000.0f

typedef _Float16 half_t;
typedef __attribute__((ext_vector_type(16))) _Float16 v16h;
typedef __attribute__((ext_vector_type(8)))  float    v8f;

#define WMMA_F16(a, b, c) \
  __builtin_amdgcn_wmma_f32_16x16x32_f16(false, (a), false, (b), (short)0, (c), false, false)

// CDNA5 async global->LDS copy (ASYNCcnt-tracked), 16B per lane.
// GVS mode: mem = SADDR(64b sgpr) + VADDR(u32 byte offset); LDS dst from VGPR.
#define ASYNC_LDS_B128(ldsaddr, goff, sbase)                                   \
  asm volatile("global_load_async_to_lds_b128 %0, %1, %2"                      \
               :: "v"(ldsaddr), "v"(goff), "s"(sbase) : "memory")

#define WAIT_ASYNC(n) asm volatile("s_wait_asynccnt %0" :: "i"(n) : "memory")

__device__ __forceinline__ unsigned lds_addr_of(const void* p) {
  return (unsigned)(unsigned long long)p;   // low 32 bits of generic addr = LDS offset
}

__device__ __forceinline__ v8f zero8() {
  v8f z = {0.f, 0.f, 0.f, 0.f, 0.f, 0.f, 0.f, 0.f};
  return z;
}

// A-fragment (16x32 f16, MxK): lane M=lane&15, g=lane>>4; half h -> K=((h&8)<<1)+8g+(h&7)
__device__ __forceinline__ v16h load_a_frag(const half_t* base, int ld) {
  const int lane = threadIdx.x & 31;
  const half_t* p = base + (size_t)(lane & 15) * ld + ((lane >> 4) << 3);
  v16h a;
#pragma unroll
  for (int h = 0; h < 16; ++h) a[h] = p[((h & 8) << 1) + (h & 7)];
  return a;
}
__device__ __forceinline__ v16h load_a_frag_f32(const float* base, int ld) {
  const int lane = threadIdx.x & 31;
  const float* p = base + (size_t)(lane & 15) * ld + ((lane >> 4) << 3);
  v16h a;
#pragma unroll
  for (int h = 0; h < 16; ++h) a[h] = (half_t)p[((h & 8) << 1) + (h & 7)];
  return a;
}

// B-fragment (32x16 f16, KxN) from 16 rows (one per N) of K-contiguous data:
// lane N=lane&15, g=lane>>4; half h -> K=((h&8)<<1)+8g+(h&7)
__device__ __forceinline__ v16h load_b_frag(const half_t* base, int ld) {
  const int lane = threadIdx.x & 31;
  const half_t* p = base + (size_t)(lane & 15) * ld + ((lane >> 4) << 3);
  v16h a;
#pragma unroll
  for (int h = 0; h < 16; ++h) a[h] = p[((h & 8) << 1) + (h & 7)];
  return a;
}
__device__ __forceinline__ v16h load_b_frag_f32(const float* base, int ld) {
  const int lane = threadIdx.x & 31;
  const float* p = base + (size_t)(lane & 15) * ld + ((lane >> 4) << 3);
  v16h a;
#pragma unroll
  for (int h = 0; h < 16; ++h) a[h] = (half_t)p[((h & 8) << 1) + (h & 7)];
  return a;
}

// ---------------------------------------------------------------------------
// GEMM: Y[R,512] = X[R,512] @ W[512,512]^T. 128 thr (4 waves), 64x64 tile.
// X/W tiles async-DMA'd (double-buffered) into LDS as raw f32 (or f16 for
// A_HALF); converted to f16 inside the fragment loaders.
// ---------------------------------------------------------------------------
template <bool A_HALF, bool OUT_HALF, bool OUT_TRANS>
__global__ void gemm_xwt(const float* __restrict__ Af, const half_t* __restrict__ Ah,
                         const float* __restrict__ W,
                         float* __restrict__ Yf, half_t* __restrict__ Yh,
                         int rowsPerBatch) {
  __shared__ __align__(16) float Xs[2][64 * 36];   // f32 stride 36 (144B) / f16 stride 40 (80B)
  __shared__ __align__(16) float Ws[2][64 * 36];
  const int rb0 = blockIdx.x * 64;
  const int cb0 = blockIdx.y * 64;
  const int tid  = threadIdx.x;
  const int wave = tid >> 5;
  const int lane = tid & 31;
  const int m = lane & 15, g = lane >> 4;

  v8f acc[4] = {zero8(), zero8(), zero8(), zero8()};

  // ----- staging: one K-chunk (32 wide) of X and W tiles into buffer `bf` --
#define GEMM_STAGE(bf, kt_) do {                                               \
    _Pragma("unroll")                                                          \
    for (int q2 = 0; q2 < 4; ++q2) {        /* W: 512 chunks / 128 thr */      \
      const int c = tid + q2 * 128;                                            \
      const int r = c >> 3, c4 = (c & 7) * 4;                                  \
      unsigned l = lds_addr_of(&Ws[bf][r * 36 + c4]);                          \
      unsigned go = (unsigned)(((cb0 + r) * C_ + (kt_) + c4) * sizeof(float)); \
      ASYNC_LDS_B128(l, go, W);                                                \
    }                                                                          \
    if (A_HALF) {                                                              \
      _Pragma("unroll")                                                        \
      for (int q2 = 0; q2 < 2; ++q2) {      /* X f16: 256 chunks */            \
        const int c = tid + q2 * 128;                                          \
        const int r = c >> 2, c8 = (c & 3) * 8;                                \
        unsigned l = lds_addr_of((const half_t*)&Xs[bf][0] + r * 40 + c8);     \
        unsigned go = (unsigned)(((rb0 + r) * C_ + (kt_) + c8) * sizeof(half_t)); \
        ASYNC_LDS_B128(l, go, Ah);                                             \
      }                                                                        \
    } else {                                                                   \
      _Pragma("unroll")                                                        \
      for (int q2 = 0; q2 < 4; ++q2) {      /* X f32: 512 chunks */            \
        const int c = tid + q2 * 128;                                          \
        const int r = c >> 3, c4 = (c & 7) * 4;                                \
        unsigned l = lds_addr_of(&Xs[bf][r * 36 + c4]);                        \
        unsigned go = (unsigned)(((rb0 + r) * C_ + (kt_) + c4) * sizeof(float)); \
        ASYNC_LDS_B128(l, go, Af);                                             \
      }                                                                        \
    }                                                                          \
  } while (0)

  GEMM_STAGE(0, 0);
  int cur = 0;
  for (int kt = 0; kt < C_; kt += 32, cur ^= 1) {
    if (kt + 32 < C_) {
      GEMM_STAGE(cur ^ 1, kt + 32);
      WAIT_ASYNC(A_HALF ? 6 : 8);   // leave only the newly-issued ops pending
    } else {
      WAIT_ASYNC(0);
    }
    __syncthreads();
    v16h a;
    if (A_HALF) a = load_a_frag((const half_t*)&Xs[cur][0] + wave * 16 * 40, 40);
    else        a = load_a_frag_f32(&Xs[cur][wave * 16 * 36], 36);
#pragma unroll
    for (int nt = 0; nt < 4; ++nt) {
      v16h b = load_b_frag_f32(&Ws[cur][nt * 16 * 36], 36);
      acc[nt] = WMMA_F16(a, b, acc[nt]);
    }
    __syncthreads();
  }
#undef GEMM_STAGE

#pragma unroll
  for (int nt = 0; nt < 4; ++nt) {
#pragma unroll
    for (int r = 0; r < 8; ++r) {
      const int row = rb0 + wave * 16 + r + 8 * g;
      const int col = cb0 + nt * 16 + m;
      const float v = acc[nt][r];
      if (OUT_TRANS) {
        const int bb = row / rowsPerBatch, mm = row % rowsPerBatch;
        const int hh = col >> 6, dd = col & 63;
        Yh[(((size_t)bb * H_ + hh) * HD_ + dd) * rowsPerBatch + mm] = (half_t)v;
      } else if (OUT_HALF) {
        Yh[(size_t)row * C_ + col] = (half_t)v;
      } else {
        Yf[(size_t)row * C_ + col] = v;
      }
    }
  }
}

// ---------------------------------------------------------------------------
// Row argmax of A @ B^T over a head slice; B tiles async-staged to LDS and
// shared by all 4 waves. First-occurrence tie-break (jnp.argmax semantics).
// ---------------------------------------------------------------------------
__global__ void row_argmax(const half_t* __restrict__ Aq, const half_t* __restrict__ Bk,
                           int* __restrict__ outIdx, int RA, int RB) {
  __shared__ __align__(16) half_t Bs[2][32 * 72];   // 32 kv rows x 64 d (stride 72)
  const int tid = threadIdx.x;
  const int wave = tid >> 5, lane = tid & 31;
  const int m = lane & 15, g = lane >> 4;
  const int tilesPerHead = RA >> 6;
  const int bh = blockIdx.x / tilesPerHead;
  const int tile = blockIdx.x % tilesPerHead;
  const int b = bh / H_, h = bh % H_;
  const int r0 = tile * 64 + wave * 16;

  const half_t* abase = Aq + ((size_t)b * RA + r0) * C_ + h * HD_;
  v16h a0 = load_a_frag(abase, C_);
  v16h a1 = load_a_frag(abase + 32, C_);
  const half_t* bbase = Bk + (size_t)b * RB * C_ + h * HD_;

#define ARG_STAGE(bf, mt_) do {                                                \
    _Pragma("unroll")                                                          \
    for (int q2 = 0; q2 < 2; ++q2) {       /* 256 chunks / 128 thr */          \
      const int c = tid + q2 * 128;                                            \
      const int r = c >> 3, c8 = (c & 7) * 8;                                  \
      unsigned l = lds_addr_of(&Bs[bf][r * 72 + c8]);                          \
      unsigned go = (unsigned)((((mt_) + r) * C_ + c8) * sizeof(half_t));      \
      ASYNC_LDS_B128(l, go, bbase);                                            \
    }                                                                          \
  } while (0)

  float bv[8];
  int   bi[8];
#pragma unroll
  for (int r = 0; r < 8; ++r) { bv[r] = -3.0e38f; bi[r] = 0; }

  ARG_STAGE(0, 0);
  int cur = 0;
  for (int mt = 0; mt < RB; mt += 32, cur ^= 1) {
    if (mt + 32 < RB) { ARG_STAGE(cur ^ 1, mt + 32); WAIT_ASYNC(2); }
    else              { WAIT_ASYNC(0); }
    __syncthreads();
#pragma unroll
    for (int s = 0; s < 2; ++s) {
      v16h b0 = load_b_frag(&Bs[cur][s * 16 * 72], 72);
      v16h b1 = load_b_frag(&Bs[cur][s * 16 * 72 + 32], 72);
      v8f sc = zero8();
      sc = WMMA_F16(a0, b0, sc);
      sc = WMMA_F16(a1, b1, sc);
      const int col = mt + s * 16 + m;
#pragma unroll
      for (int r = 0; r < 8; ++r) {
        const float v = sc[r];
        if (v > bv[r]) { bv[r] = v; bi[r] = col; }   // within-lane cols ascend
      }
    }
    __syncthreads();
  }
#undef ARG_STAGE

#pragma unroll
  for (int mask = 1; mask < 16; mask <<= 1) {
#pragma unroll
    for (int r = 0; r < 8; ++r) {
      const float ov = __shfl_xor(bv[r], mask, 32);
      const int   oi = __shfl_xor(bi[r], mask, 32);
      if (ov > bv[r] || (ov == bv[r] && oi < bi[r])) { bv[r] = ov; bi[r] = oi; }
    }
  }
  if (m == 0) {
#pragma unroll
    for (int r = 0; r < 8; ++r)
      outIdx[(size_t)bh * RA + r0 + r + 8 * g] = bi[r];
  }
}

// ---------------------------------------------------------------------------
// Cycle-consistency mask -> additive attention bias per (b,h,m)
// ---------------------------------------------------------------------------
__global__ void bias_kernel(const int* __restrict__ q2k, const int* __restrict__ k2q,
                            const float* __restrict__ svm, const int* __restrict__ smask,
                            float* __restrict__ bias) {
  const int i = blockIdx.x * blockDim.x + threadIdx.x;   // over B*H*NS
  if (i >= B_ * H_ * NS_) return;
  const int mm = i % NS_;
  const int bh = i / NS_;
  const int b = bh / H_;
  const int i1 = k2q[i];
  const int re = q2k[(size_t)bh * N_ + i1];
  const bool asso = smask[b * NS_ + mm] == smask[b * NS_ + re];
  const float valid = asso ? svm[b * NS_ + mm] : 1.0f;
  bias[i] = valid * NEG_;
}

// ---------------------------------------------------------------------------
// Flash attention with bias. 64 q-rows/block (16/wave); K and V tiles
// async-DMA'd into double-buffered LDS and shared by all 4 waves.
// vt layout is [B,H,HD,NS] so V fragments are K-contiguous.
// ---------------------------------------------------------------------------
__global__ void flash_attn(const half_t* __restrict__ qh, const half_t* __restrict__ kh,
                           const half_t* __restrict__ vt, const float* __restrict__ bias,
                           half_t* __restrict__ xh) {
  __shared__ __align__(16) half_t Ks[2][32 * 72];   // 32 kv x 64 d (stride 72)
  __shared__ __align__(16) half_t Vs[2][64 * 40];   // 64 d  x 32 kv (stride 40)
  __shared__ __align__(16) half_t Pl[4][16 * 40];   // per-wave P tiles
  const int tid = threadIdx.x;
  const int wave = tid >> 5, lane = tid & 31;
  const int m = lane & 15, g = lane >> 4;
  const int tilesPerHead = N_ >> 6;
  const int bh = blockIdx.x / tilesPerHead;
  const int tile = blockIdx.x % tilesPerHead;
  const int b = bh / H_, h = bh % H_;
  const int r0 = tile * 64 + wave * 16;

  const half_t* qbase = qh + ((size_t)b * N_ + r0) * C_ + h * HD_;
  v16h a0 = load_a_frag(qbase, C_);
  v16h a1 = load_a_frag(qbase + 32, C_);
  const half_t* kbase = kh + (size_t)b * NS_ * C_ + h * HD_;
  const half_t* vbase = vt + ((size_t)b * H_ + h) * HD_ * NS_;
  const float*  bp = bias + (size_t)bh * NS_;

#define FLASH_STAGE(bf, mt_) do {                                              \
    _Pragma("unroll")                                                          \
    for (int q2 = 0; q2 < 2; ++q2) {                                           \
      const int c = tid + q2 * 128;                                            \
      /* K chunk: row=c>>3 (kv), col=(c&7)*8 (d) */                            \
      const int kr = c >> 3, kc = (c & 7) * 8;                                 \
      unsigned lk = lds_addr_of(&Ks[bf][kr * 72 + kc]);                        \
      unsigned gk = (unsigned)((((mt_) + kr) * C_ + kc) * sizeof(half_t));     \
      ASYNC_LDS_B128(lk, gk, kbase);                                           \
      /* V chunk: row=c>>2 (d), col=(c&3)*8 (kv) */                            \
      const int vr = c >> 2, vc = (c & 3) * 8;                                 \
      unsigned lv = lds_addr_of(&Vs[bf][vr * 40 + vc]);                        \
      unsigned gv = (unsigned)((vr * NS_ + (mt_) + vc) * sizeof(half_t));      \
      ASYNC_LDS_B128(lv, gv, vbase);                                           \
    }                                                                          \
  } while (0)

  v8f o0 = zero8(), o1 = zero8(), o2 = zero8(), o3 = zero8();
  float rmax[8], rsum[8];
#pragma unroll
  for (int r = 0; r < 8; ++r) { rmax[r] = -3.0e38f; rsum[r] = 0.f; }
  half_t* P = Pl[wave];

  FLASH_STAGE(0, 0);
  int cur = 0;
  for (int mt = 0; mt < NS_; mt += 32, cur ^= 1) {
    if (mt + 32 < NS_) { FLASH_STAGE(cur ^ 1, mt + 32); WAIT_ASYNC(4); }
    else               { WAIT_ASYNC(0); }
    __syncthreads();

    v16h k00 = load_b_frag(&Ks[cur][0], 72);
    v16h k01 = load_b_frag(&Ks[cur][32], 72);
    v16h k10 = load_b_frag(&Ks[cur][16 * 72], 72);
    v16h k11 = load_b_frag(&Ks[cur][16 * 72 + 32], 72);
    v8f s0 = zero8(), s1 = zero8();
    s0 = WMMA_F16(a0, k00, s0);
    s0 = WMMA_F16(a1, k01, s0);
    s1 = WMMA_F16(a0, k10, s1);
    s1 = WMMA_F16(a1, k11, s1);

    const float bias0 = bp[mt + m];
    const float bias1 = bp[mt + 16 + m];
    float cm[8];
#pragma unroll
    for (int r = 0; r < 8; ++r) {
      s0[r] = s0[r] * SCALE_ + bias0;
      s1[r] = s1[r] * SCALE_ + bias1;
      cm[r] = fmaxf(s0[r], s1[r]);
    }
#pragma unroll
    for (int mask = 1; mask < 16; mask <<= 1) {
#pragma unroll
      for (int r = 0; r < 8; ++r) cm[r] = fmaxf(cm[r], __shfl_xor(cm[r], mask, 32));
    }
    float alpha[8], p0[8], p1[8], cs[8];
#pragma unroll
    for (int r = 0; r < 8; ++r) {
      const float nm = fmaxf(rmax[r], cm[r]);
      alpha[r] = __expf(rmax[r] - nm);
      rmax[r]  = nm;
      p0[r] = __expf(s0[r] - nm);
      p1[r] = __expf(s1[r] - nm);
      cs[r] = p0[r] + p1[r];
    }
#pragma unroll
    for (int mask = 1; mask < 16; mask <<= 1) {
#pragma unroll
      for (int r = 0; r < 8; ++r) cs[r] += __shfl_xor(cs[r], mask, 32);
    }
#pragma unroll
    for (int r = 0; r < 8; ++r) {
      rsum[r] = rsum[r] * alpha[r] + cs[r];
      o0[r] *= alpha[r]; o1[r] *= alpha[r]; o2[r] *= alpha[r]; o3[r] *= alpha[r];
      P[(r + 8 * g) * 40 + m]      = (half_t)p0[r];
      P[(r + 8 * g) * 40 + 16 + m] = (half_t)p1[r];
    }
    v16h ap = load_a_frag(P, 40);
    v16h v0 = load_b_frag(&Vs[cur][0 * 16 * 40], 40);
    v16h v1 = load_b_frag(&Vs[cur][1 * 16 * 40], 40);
    v16h v2 = load_b_frag(&Vs[cur][2 * 16 * 40], 40);
    v16h v3 = load_b_frag(&Vs[cur][3 * 16 * 40], 40);
    o0 = WMMA_F16(ap, v0, o0);
    o1 = WMMA_F16(ap, v1, o1);
    o2 = WMMA_F16(ap, v2, o2);
    o3 = WMMA_F16(ap, v3, o3);
    __syncthreads();
  }
#undef FLASH_STAGE

  half_t* xb = xh + ((size_t)b * N_ + r0) * C_ + h * HD_;
#pragma unroll
  for (int r = 0; r < 8; ++r) {
    const float inv = 1.0f / rsum[r];
    const int row = r + 8 * g;
    xb[(size_t)row * C_ + 0  + m] = (half_t)(o0[r] * inv);
    xb[(size_t)row * C_ + 16 + m] = (half_t)(o1[r] * inv);
    xb[(size_t)row * C_ + 32 + m] = (half_t)(o2[r] * inv);
    xb[(size_t)row * C_ + 48 + m] = (half_t)(o3[r] * inv);
  }
}

// ---------------------------------------------------------------------------
extern "C" void kernel_launch(void* const* d_in, const int* in_sizes, int n_in,
                              void* d_out, int out_size, void* d_ws, size_t ws_size,
                              hipStream_t stream) {
  (void)in_sizes; (void)n_in; (void)out_size; (void)ws_size;
  const float* q     = (const float*)d_in[0];
  const float* k     = (const float*)d_in[1];
  const float* v     = (const float*)d_in[2];
  const float* svm   = (const float*)d_in[3];
  const int*   smask = (const int*)d_in[4];
  const float* Wq    = (const float*)d_in[5];
  const float* Wk    = (const float*)d_in[6];
  const float* Wv    = (const float*)d_in[7];
  const float* Wproj = (const float*)d_in[8];
  float* out = (float*)d_out;

  char* ws = (char*)d_ws;
  size_t off = 0;
  half_t* qh  = (half_t*)(ws + off); off += (size_t)B_ * N_  * C_ * sizeof(half_t);
  half_t* kh  = (half_t*)(ws + off); off += (size_t)B_ * NS_ * C_ * sizeof(half_t);
  half_t* vt  = (half_t*)(ws + off); off += (size_t)B_ * NS_ * C_ * sizeof(half_t);
  half_t* xh  = (half_t*)(ws + off); off += (size_t)B_ * N_  * C_ * sizeof(half_t);
  int*    q2k = (int*)(ws + off);    off += (size_t)B_ * H_ * N_  * sizeof(int);
  int*    k2q = (int*)(ws + off);    off += (size_t)B_ * H_ * NS_ * sizeof(int);
  float*  bias= (float*)(ws + off);  off += (size_t)B_ * H_ * NS_ * sizeof(float);

  const dim3 gg(B_ * N_ / 64, C_ / 64);   // (64, 8)

  // Projections (f32 in, f16 out; V stored head-transposed)
  gemm_xwt<false, true, false><<<gg, 128, 0, stream>>>(q, nullptr, Wq, nullptr, qh, N_);
  gemm_xwt<false, true, false><<<gg, 128, 0, stream>>>(k, nullptr, Wk, nullptr, kh, NS_);
  gemm_xwt<false, true, true ><<<gg, 128, 0, stream>>>(v, nullptr, Wv, nullptr, vt, NS_);

  // Cycle-consistency argmaxes
  row_argmax<<<B_ * H_ * N_  / 64, 128, 0, stream>>>(qh, kh, q2k, N_,  NS_);
  row_argmax<<<B_ * H_ * NS_ / 64, 128, 0, stream>>>(kh, qh, k2q, NS_, N_);

  // Mask -> additive bias
  bias_kernel<<<(B_ * H_ * NS_ + 255) / 256, 256, 0, stream>>>(q2k, k2q, svm, smask, bias);

  // Flash attention with bias
  flash_attn<<<B_ * H_ * N_ / 64, 128, 0, stream>>>(qh, kh, vt, bias, xh);

  // Output projection (f16 A, f32 out)
  gemm_xwt<true, false, false><<<gg, 128, 0, stream>>>(nullptr, xh, Wproj, out, nullptr, N_);
}
